// GSVDLayer_34248069218802
// MI455X (gfx1250) — matrix operand order repaired and checked by reading, not compiled, not run on hardware
//
#include <hip/hip_runtime.h>
#include <hip/hip_bf16.h>
#include <stdint.h>

typedef __attribute__((ext_vector_type(16))) __bf16 v16bf;
typedef __attribute__((ext_vector_type(8)))  __bf16 v8bf;
typedef __attribute__((ext_vector_type(4)))  __bf16 v4bf;
typedef __attribute__((ext_vector_type(8)))  float  v8f;
typedef __attribute__((ext_vector_type(4)))  float  v4f;

#define D_IN  4096
#define D_OUT 4096
#define RNK   4096
#define MSEQ  8192   // B*SEQ = 4*2048

// ---------------- f32 -> bf16 converter (vector x4) ----------------
__global__ void cvt_f32_bf16(const float* __restrict__ in,
                             __bf16* __restrict__ out, int n) {
    int i = (blockIdx.x * blockDim.x + threadIdx.x) * 4;
    if (i + 3 < n) {
        v4f v = *(const v4f*)(in + i);
        v4bf o;
        o[0] = (__bf16)v.x; o[1] = (__bf16)v.y;
        o[2] = (__bf16)v.z; o[3] = (__bf16)v.w;
        *(v4bf*)(out + i) = o;
    }
}

// ------- VsT[i][r] = bf16(S[r] * Vh[r][i])  (tiled LDS transpose) -------
__global__ void scale_transpose(const float* __restrict__ Vh,
                                const float* __restrict__ S,
                                __bf16* __restrict__ VsT) {
    __shared__ float tile[32][33];
    int bi = blockIdx.x * 32;   // D_IN offset
    int br = blockIdx.y * 32;   // R offset
    int tx = threadIdx.x;       // 0..31
    int ty = threadIdx.y;       // 0..7
    #pragma unroll
    for (int j = 0; j < 32; j += 8) {
        int r = br + ty + j;
        tile[ty + j][tx] = S[r] * Vh[(size_t)r * D_IN + bi + tx];
    }
    __syncthreads();
    #pragma unroll
    for (int j = 0; j < 32; j += 8) {
        int i = bi + ty + j;
        VsT[(size_t)i * RNK + br + tx] = (__bf16)tile[tx][ty + j];
    }
}

// ---------------- NT GEMM: C[m][n] = sum_k A[m][k]*B[n][k] ----------------
// A: [M][K] bf16 row-major, B: [N][K] bf16 row-major, C: [M][N] (bf16 or f32)
// Block tile 256x128x32, 256 threads = 8 waves (4x2), wave tile 64x64.
// Double-buffered LDS fed by GLOBAL_LOAD_ASYNC_TO_LDS_B128 in saddr (GVS)
// form: loop-invariant per-lane voffsets, uniform SGPR base advanced by SALU.
template <typename CT>
__global__ __launch_bounds__(256)
void gemm_bf16_nt(const __bf16* __restrict__ A, const __bf16* __restrict__ B,
                  CT* __restrict__ C, int M, int N, int K) {
    constexpr int BM = 256, BN = 128, BK = 32, PAD = 8;
    constexpr int LDK  = BK + PAD;         // 40 bf16 = 80 B row stride
    constexpr int ASZ  = BM * LDK;         // 10240 elems
    constexpr int BSZ  = BN * LDK;         //  5120 elems
    constexpr int BUFE = ASZ + BSZ;        // elems per buffer
    constexpr uint32_t BUFB = BUFE * 2;    // bytes per buffer
    __shared__ __bf16 sm[2 * BUFE];        // 60 KB

    const int tid    = threadIdx.x;
    const int lane   = tid & 31;
    const int wid    = tid >> 5;
    const int lane16 = lane & 15;
    const int half   = lane >> 4;          // 0 or 1

    const int bm = blockIdx.y * BM;
    const int bn = blockIdx.x * BN;
    const int wm = (wid >> 1) * 64;        // wave row offset: 0/64/128/192
    const int wn = (wid & 1)  * 64;        // wave col offset: 0/64

    // ---- async-copy slots: 4 A-chunks + 2 B-chunks of 16 B per thread ----
    // A: 1024 chunks (row=c>>2, kchunk=(c&3)*8); B: 512 chunks.
    uint32_t voA[4], loA[2][4];            // loop-invariant
    uint32_t voB[2], loB[2][2];
    #pragma unroll
    for (int j = 0; j < 4; ++j) {
        int c = tid + j * 256;
        int r = c >> 2, kc = (c & 3) << 3;
        voA[j]    = (uint32_t)(((size_t)r * K + kc) * 2);
        loA[0][j] = (uint32_t)(uintptr_t)(&sm[r * LDK + kc]);
        loA[1][j] = loA[0][j] + BUFB;
    }
    #pragma unroll
    for (int j = 0; j < 2; ++j) {
        int c = tid + j * 256;
        int r = c >> 2, kc = (c & 3) << 3;
        voB[j]    = (uint32_t)(((size_t)r * K + kc) * 2);
        loB[0][j] = (uint32_t)(uintptr_t)(&sm[ASZ + r * LDK + kc]);
        loB[1][j] = loB[0][j] + BUFB;
    }

    // uniform tile bases (SGPRs); advance by BK*2 bytes per issued tile
    const char* gA = (const char*)A + (size_t)bm * K * 2;
    const char* gB = (const char*)B + (size_t)bn * K * 2;

    auto issue = [&](int bufSel) {         // bufSel is a literal at call sites
        #pragma unroll
        for (int j = 0; j < 4; ++j)
            asm volatile("global_load_async_to_lds_b128 %0, %1, %2"
                         :: "v"(loA[bufSel][j]), "v"(voA[j]), "s"(gA)
                         : "memory");
        #pragma unroll
        for (int j = 0; j < 2; ++j)
            asm volatile("global_load_async_to_lds_b128 %0, %1, %2"
                         :: "v"(loB[bufSel][j]), "v"(voB[j]), "s"(gB)
                         : "memory");
        gA += BK * 2;
        gB += BK * 2;
    };

    v8f acc[4][4] = {};
    union Frag { v16bf v; v8bf h[2]; };

    auto compute = [&](const __bf16* sA) {
        const __bf16* sB = sA + ASZ;
        // B fragments: lane = column (lane16), 16 contiguous K at half*16
        Frag bfr[4];
        #pragma unroll
        for (int jn = 0; jn < 4; ++jn) {
            const __bf16* bp = sB + (wn + jn * 16 + lane16) * LDK + half * 16;
            bfr[jn].h[0] = *(const v8bf*)(bp);
            bfr[jn].h[1] = *(const v8bf*)(bp + 8);
        }
        // A fragments + 16 WMMAs
        #pragma unroll
        for (int im = 0; im < 4; ++im) {
            const __bf16* ap = sA + (wm + im * 16 + lane16) * LDK + half * 16;
            Frag af;
            af.h[0] = *(const v8bf*)(ap);        // K = half*16 .. +7
            af.h[1] = *(const v8bf*)(ap + 8);    // K = half*16+8 .. +15
            #pragma unroll
            for (int jn = 0; jn < 4; ++jn) {
                acc[im][jn] = __builtin_amdgcn_wmma_f32_16x16x32_bf16(
                    false, af.v, false, bfr[jn].v, (short)0, acc[im][jn],
                    false, false);
            }
        }
    };

    #define ASYNC_JOIN() do {                                   \
        asm volatile("s_wait_asynccnt 0x0" ::: "memory");       \
        __syncthreads();                                        \
    } while (0)

    // prologue: tile 0 -> buffer 0
    issue(0);
    ASYNC_JOIN();

    const int NT = K / BK;                 // 128 (even)
    for (int t = 0; t < NT; t += 2) {
        issue(1);                          // DMA tile t+1 (always exists)
        compute(sm);                       // tile t from buffer 0
        ASYNC_JOIN();
        if (t + 2 < NT) issue(0);          // DMA tile t+2
        compute(sm + BUFE);                // tile t+1 from buffer 1
        ASYNC_JOIN();
    }
    #undef ASYNC_JOIN

    // ---- epilogue: VGPR r -> row (half*8 + r), col = lane16 ----
    #pragma unroll
    for (int im = 0; im < 4; ++im) {
        #pragma unroll
        for (int jn = 0; jn < 4; ++jn) {
            int col = bn + wn + jn * 16 + lane16;
            #pragma unroll
            for (int r = 0; r < 8; ++r) {
                int row = bm + wm + im * 16 + half * 8 + r;
                C[(size_t)row * N + col] = (CT)acc[im][jn][r];
            }
        }
    }
}

// ---------------- host-side orchestration ----------------
extern "C" void kernel_launch(void* const* d_in, const int* in_sizes, int n_in,
                              void* d_out, int out_size, void* d_ws, size_t ws_size,
                              hipStream_t stream) {
    const float* x  = (const float*)d_in[0];  // [4,2048,4096]
    const float* U  = (const float*)d_in[1];  // [4096,4096]
    const float* S  = (const float*)d_in[2];  // [4096]
    const float* Vh = (const float*)d_in[3];  // [4096,4096]
    float* out = (float*)d_out;               // [4,2048,4096] f32

    // workspace layout (needs 160 MB)
    char* ws = (char*)d_ws;
    __bf16* xbf = (__bf16*)(ws);                            // 64 MB
    __bf16* Ubf = (__bf16*)(ws + ((size_t)64  << 20));      // 32 MB
    __bf16* VsT = (__bf16*)(ws + ((size_t)96  << 20));      // 32 MB
    __bf16* Wbf = (__bf16*)(ws + ((size_t)128 << 20));      // 32 MB

    const int nx = MSEQ * D_IN;    // 33.5M
    const int nu = D_OUT * RNK;    // 16.8M
    cvt_f32_bf16<<<nx / 1024, 256, 0, stream>>>(x, xbf, nx);
    cvt_f32_bf16<<<nu / 1024, 256, 0, stream>>>(U, Ubf, nu);

    scale_transpose<<<dim3(D_IN / 32, RNK / 32), dim3(32, 8), 0, stream>>>(Vh, S, VsT);

    // W[o][i] = sum_r Ubf[o][r] * VsT[i][r]   (M=D_OUT, N=D_IN, K=R)
    gemm_bf16_nt<__bf16><<<dim3(D_IN / 128, D_OUT / 256), 256, 0, stream>>>(
        Ubf, VsT, Wbf, D_OUT, D_IN, RNK);

    // out[m][o] = sum_i xbf[m][i] * Wbf[o][i]  (M=8192, N=D_OUT, K=D_IN)
    gemm_bf16_nt<float><<<dim3(D_OUT / 128, MSEQ / 256), 256, 0, stream>>>(
        xbf, Wbf, out, MSEQ, D_OUT, D_IN);
}
